// ImprovedPointNetExtractorNEW_86268713107569
// MI455X (gfx1250) — matrix operand order repaired and checked by reading, not compiled, not run on hardware
//
#include <hip/hip_runtime.h>
#include <hip/hip_bf16.h>

#define BB   8
#define NN   4096
#define KNB  10
#define EPSV 1e-5f

typedef __bf16 bf16;
typedef __attribute__((ext_vector_type(16))) __bf16 v16bf;
typedef __attribute__((ext_vector_type(8)))  __bf16 v8bf;
typedef __attribute__((ext_vector_type(8)))  float  v8f;

// ---------------- WMMA fragment helpers (CDNA5 16x16x32 bf16 layout) ----------------
// A (16x32, row-major source): lane L holds row M=L&15; half=L>>4.
//   elems 0..7  -> K = kk + 8*half + t
//   elems 8..15 -> K = kk + 16 + 8*half + t
static __device__ __forceinline__ v16bf load_a_frag(const bf16* p) {
    v8bf lo = *(const v8bf*)p;          // K = kk+8*half .. +7
    v8bf hi = *(const v8bf*)(p + 16);   // K = kk+16+8*half .. +7
    v16bf a;
#pragma unroll
    for (int t = 0; t < 8; ++t) { a[t] = lo[t]; a[8 + t] = hi[t]; }
    return a;
}
// B (32x16): lane L holds column N=L&15; elems 0..7 -> K = kk+8*half+t, elems 8..15 -> K = kk+16+8*half+t.
static __device__ __forceinline__ v16bf load_b_frag(const bf16* p, int ld) {
    v16bf b;
#pragma unroll
    for (int t = 0; t < 8; ++t) { b[t] = p[(long)t * ld]; b[8 + t] = p[(long)(16 + t) * ld]; }
    return b;
}

// ---------------- utility kernels ----------------
__global__ void k_zero(float* __restrict__ p, long n) {
    long stride = (long)gridDim.x * blockDim.x;
    for (long i = (long)blockIdx.x * blockDim.x + threadIdx.x; i < n; i += stride) p[i] = 0.f;
}

__global__ void k_cvt(const float* __restrict__ s, bf16* __restrict__ d, int n) {
    int i = blockIdx.x * blockDim.x + threadIdx.x;
    if (i < n) d[i] = (bf16)s[i];
}

// ---------------- KNN: top-11 nearest (d2 = sn + sm - 2*dot), LDS-tiled ----------------
__global__ __launch_bounds__(256) void k_knn(const float* __restrict__ x, int* __restrict__ idx11) {
    __shared__ float s_cx[256], s_cy[256], s_cz[256], s_sq[256];
    const int b = blockIdx.y;
    const int n = blockIdx.x * 256 + threadIdx.x;
    const float* xb = x + (long)b * NN * 5;
    const float cx = xb[n * 5 + 0], cy = xb[n * 5 + 1], cz = xb[n * 5 + 2];
    const float sn = cx * cx + cy * cy + cz * cz;
    float bd[11]; int bi[11];
#pragma unroll
    for (int t = 0; t < 11; ++t) { bd[t] = 3.4e38f; bi[t] = 0; }
    for (int ct = 0; ct < NN; ct += 256) {
        const int m = ct + threadIdx.x;
        s_cx[threadIdx.x] = xb[m * 5 + 0];
        s_cy[threadIdx.x] = xb[m * 5 + 1];
        s_cz[threadIdx.x] = xb[m * 5 + 2];
        s_sq[threadIdx.x] = s_cx[threadIdx.x] * s_cx[threadIdx.x] +
                            s_cy[threadIdx.x] * s_cy[threadIdx.x] +
                            s_cz[threadIdx.x] * s_cz[threadIdx.x];
        __syncthreads();
        for (int mm = 0; mm < 256; ++mm) {
            const float dot = cx * s_cx[mm] + cy * s_cy[mm] + cz * s_cz[mm];
            const float d = sn + s_sq[mm] - 2.0f * dot;
            if (d < bd[10]) {
                float cd = d; int ci = ct + mm;
#pragma unroll
                for (int p = 0; p < 11; ++p) {
                    if (cd < bd[p]) {
                        float td = bd[p]; int ti = bi[p];
                        bd[p] = cd; bi[p] = ci; cd = td; ci = ti;
                    }
                }
            }
        }
        __syncthreads();
    }
    int* op = idx11 + ((long)b * NN + n) * 11;
#pragma unroll
    for (int t = 0; t < 11; ++t) op[t] = bi[t];
}

// ---------------- layer-1 edge feature helper ----------------
static __device__ __forceinline__ void build_e1(const float* __restrict__ x, int b, int n, int j, float e[10]) {
    const float* xn = x + ((long)b * NN + n) * 5;
    const float* xj = x + ((long)b * NN + j) * 5;
    e[0] = xj[0] - xn[0]; e[1] = xj[1] - xn[1]; e[2] = xj[2] - xn[2];
    e[3] = xn[0];         e[4] = xn[1];         e[5] = xn[2];
    e[6] = xj[3] - xn[3]; e[7] = xj[4] - xn[4];
    e[8] = xn[3];         e[9] = xn[4];
}

// pass 1: per-channel sum / sumsq of y1 = e1 @ W1
__global__ __launch_bounds__(256) void k_l1_stats(const float* __restrict__ x, const int* __restrict__ idx11,
                                                  const float* __restrict__ W1, float* __restrict__ stats1) {
    __shared__ float s_sum[64], s_sq[64];
    const int tid = threadIdx.x;
    if (tid < 64) { s_sum[tid] = 0.f; s_sq[tid] = 0.f; }
    __syncthreads();
    const int R = blockIdx.x * 256 + tid;          // 0 .. B*N*KNB-1
    const int b = R / (NN * KNB);
    const int rem = R % (NN * KNB);
    const int n = rem / KNB, k = rem % KNB;
    const int j = idx11[((long)b * NN + n) * 11 + k];   // idx1 = first K (incl self)
    float e[10]; build_e1(x, b, n, j, e);
    for (int o = 0; o < 64; ++o) {
        float y = 0.f;
#pragma unroll
        for (int i = 0; i < 10; ++i) y = fmaf(e[i], W1[i * 64 + o], y);
        atomicAdd(&s_sum[o], y);
        atomicAdd(&s_sq[o], y * y);
    }
    __syncthreads();
    if (tid < 64) { atomicAdd(&stats1[tid], s_sum[tid]); atomicAdd(&stats1[64 + tid], s_sq[tid]); }
}

// pass 2: f1 = max_k relu(bn(y1))
__global__ __launch_bounds__(256) void k_l1_apply(const float* __restrict__ x, const int* __restrict__ idx11,
                                                  const float* __restrict__ W1, const float* __restrict__ g1,
                                                  const float* __restrict__ b1, const float* __restrict__ stats1,
                                                  float* __restrict__ f1) {
    const int P = blockIdx.x * 256 + threadIdx.x;   // point
    const int b = P / NN, n = P % NN;
    const float inv = 1.0f / (float)((long)BB * NN * KNB);
    float best[64];
#pragma unroll
    for (int o = 0; o < 64; ++o) best[o] = -3.4e38f;
    for (int k = 0; k < KNB; ++k) {
        const int j = idx11[((long)b * NN + n) * 11 + k];
        float e[10]; build_e1(x, b, n, j, e);
        for (int o = 0; o < 64; ++o) {
            float y = 0.f;
#pragma unroll
            for (int i = 0; i < 10; ++i) y = fmaf(e[i], W1[i * 64 + o], y);
            const float mean = stats1[o] * inv;
            const float var = stats1[64 + o] * inv - mean * mean;
            const float sc = g1[o] * rsqrtf(var + EPSV);
            float h = (y - mean) * sc + b1[o];
            h = fmaxf(h, 0.f);
            best[o] = fmaxf(best[o], h);
        }
    }
    float* out = f1 + (long)P * 64;
#pragma unroll
    for (int o = 0; o < 64; ++o) out[o] = best[o];
}

// ---------------- soft-edge pass 1: WMMA GEMM -> channel stats + attention logits ----------------
template <int IN, int OUT, int AOUT>
__global__ __launch_bounds__(128) void k_soft_stats(const float* __restrict__ f, const int* __restrict__ idx11,
                                                    const bf16* __restrict__ Wm, const bf16* __restrict__ Wa,
                                                    const float* __restrict__ Wb, const float* __restrict__ ba,
                                                    const float* __restrict__ bb,
                                                    float* __restrict__ stats, float* __restrict__ logits) {
    constexpr int F = IN / 2;
    constexpr int ROWS = 64;
    constexpr int LDA = IN + 8;
    __shared__ __align__(16) bf16 As[ROWS * LDA];
    __shared__ float s_sum[OUT], s_sq[OUT], s_logit[ROWS];
    __shared__ int s_j[ROWS], s_n[ROWS], s_b[ROWS];

    const int tid = threadIdx.x, lane = tid & 31, wv = tid >> 5;
    const int R0 = blockIdx.x * ROWS;

    for (int i = tid; i < OUT; i += 128) { s_sum[i] = 0.f; s_sq[i] = 0.f; }
    if (tid < ROWS) {
        const int R = R0 + tid;
        const int b = R / (NN * KNB);
        const int rem = R % (NN * KNB);
        const int n = rem / KNB, k = rem % KNB;
        s_b[tid] = b; s_n[tid] = n;
        s_j[tid] = idx11[((long)b * NN + n) * 11 + k + 1];   // skip self
        s_logit[tid] = 0.f;
    }
    __syncthreads();
    for (int e = tid; e < ROWS * IN; e += 128) {
        const int r = e / IN, i = e % IN;
        const int b = s_b[r], n = s_n[r], j = s_j[r];
        float v;
        if (i < F) v = f[((long)b * NN + j) * F + i] - f[((long)b * NN + n) * F + i];
        else       v = f[((long)b * NN + n) * F + (i - F)];
        As[r * LDA + i] = (bf16)v;
    }
    __syncthreads();

    const int half = lane >> 4;
    const int mrow = lane & 15;
    const bf16* arow = &As[(wv * 16 + mrow) * LDA + 8 * half];

    // main GEMM: y = e @ Wm, accumulate per-channel stats
    for (int ct = 0; ct < OUT / 16; ++ct) {
        v8f acc = {};
        for (int kk = 0; kk < IN; kk += 32) {
            v16bf a = load_a_frag(arow + kk);
            v16bf bm = load_b_frag(Wm + (long)(kk + 8 * half) * OUT + ct * 16 + mrow, OUT);
            acc = __builtin_amdgcn_wmma_f32_16x16x32_bf16(false, a, false, bm, (short)0, acc, false, false);
        }
        float s = 0.f, q = 0.f;
#pragma unroll
        for (int t = 0; t < 8; ++t) { const float v = acc[t]; s += v; q += v * v; }
        s += __shfl_xor(s, 16, 32);
        q += __shfl_xor(q, 16, 32);
        if (half == 0) { atomicAdd(&s_sum[ct * 16 + mrow], s); atomicAdd(&s_sq[ct * 16 + mrow], q); }
    }

    // attention GEMM: logit = sum_o Wb[o]*relu((e@Wa)[o] + ba[o]) + bb
    for (int ct = 0; ct < AOUT / 16; ++ct) {
        v8f acc = {};
        for (int kk = 0; kk < IN; kk += 32) {
            v16bf a = load_a_frag(arow + kk);
            v16bf bm = load_b_frag(Wa + (long)(kk + 8 * half) * AOUT + ct * 16 + mrow, AOUT);
            acc = __builtin_amdgcn_wmma_f32_16x16x32_bf16(false, a, false, bm, (short)0, acc, false, false);
        }
        const float wbo = Wb[ct * 16 + mrow];
        const float bao = ba[ct * 16 + mrow];
#pragma unroll
        for (int t = 0; t < 8; ++t) {
            float h = fmaxf(acc[t] + bao, 0.f);
            atomicAdd(&s_logit[wv * 16 + t + 8 * half], h * wbo);
        }
    }
    __syncthreads();
    if (tid < ROWS) logits[R0 + tid] = s_logit[tid] + bb[0];
    for (int i = tid; i < OUT; i += 128) {
        atomicAdd(&stats[i], s_sum[i]);
        atomicAdd(&stats[OUT + i], s_sq[i]);
    }
}

// ---------------- softmax over K neighbors (with temperature) ----------------
__global__ __launch_bounds__(256) void k_softmax_k(const float* __restrict__ logits, float* __restrict__ alpha) {
    const int p = blockIdx.x * 256 + threadIdx.x;   // b*N+n
    const float* l = logits + (long)p * KNB;
    float mx = -3.4e38f;
#pragma unroll
    for (int k = 0; k < KNB; ++k) mx = fmaxf(mx, l[k]);
    float ex[KNB]; float z = 0.f;
#pragma unroll
    for (int k = 0; k < KNB; ++k) { ex[k] = expf((l[k] - mx) * 5.0f); z += ex[k]; }  // 1/TAU = 5
    const float iz = 1.0f / z;
    float* a = alpha + (long)p * KNB;
#pragma unroll
    for (int k = 0; k < KNB; ++k) a[k] = ex[k] * iz;
}

// ---------------- soft-edge pass 2: WMMA GEMM -> BN/ReLU * alpha -> sum over k ----------------
template <int IN, int OUT>
__global__ __launch_bounds__(128) void k_soft_apply(const float* __restrict__ f, const int* __restrict__ idx11,
                                                    const bf16* __restrict__ Wm, const float* __restrict__ stats,
                                                    const float* __restrict__ g, const float* __restrict__ bt,
                                                    const float* __restrict__ alpha, float* __restrict__ fout) {
    constexpr int F = IN / 2;
    constexpr int ROWS = 64;
    constexpr int LDA = IN + 8;
    __shared__ __align__(16) bf16 As[ROWS * LDA];
    __shared__ float s_alpha[ROWS];
    __shared__ int s_j[ROWS], s_n[ROWS], s_b[ROWS];

    const int tid = threadIdx.x, lane = tid & 31, wv = tid >> 5;
    const int R0 = blockIdx.x * ROWS;
    const float inv = 1.0f / (float)((long)BB * NN * KNB);

    if (tid < ROWS) {
        const int R = R0 + tid;
        const int b = R / (NN * KNB);
        const int rem = R % (NN * KNB);
        const int n = rem / KNB, k = rem % KNB;
        s_b[tid] = b; s_n[tid] = n;
        s_j[tid] = idx11[((long)b * NN + n) * 11 + k + 1];
        s_alpha[tid] = alpha[R];
    }
    __syncthreads();
    for (int e = tid; e < ROWS * IN; e += 128) {
        const int r = e / IN, i = e % IN;
        const int b = s_b[r], n = s_n[r], j = s_j[r];
        float v;
        if (i < F) v = f[((long)b * NN + j) * F + i] - f[((long)b * NN + n) * F + i];
        else       v = f[((long)b * NN + n) * F + (i - F)];
        As[r * LDA + i] = (bf16)v;
    }
    __syncthreads();

    const int half = lane >> 4;
    const int mrow = lane & 15;
    const bf16* arow = &As[(wv * 16 + mrow) * LDA + 8 * half];

    for (int ct = 0; ct < OUT / 16; ++ct) {
        v8f acc = {};
        for (int kk = 0; kk < IN; kk += 32) {
            v16bf a = load_a_frag(arow + kk);
            v16bf bm = load_b_frag(Wm + (long)(kk + 8 * half) * OUT + ct * 16 + mrow, OUT);
            acc = __builtin_amdgcn_wmma_f32_16x16x32_bf16(false, a, false, bm, (short)0, acc, false, false);
        }
        const int o = ct * 16 + mrow;
        const float mean = stats[o] * inv;
        const float var = stats[OUT + o] * inv - mean * mean;
        const float sc = g[o] * rsqrtf(var + EPSV);
        const float sh = bt[o] - mean * sc;
        // 8 consecutive rows per lane: run-length compress over n before atomics
        float run = 0.f; int curn = -1, curb = 0;
#pragma unroll
        for (int t = 0; t < 8; ++t) {
            const int r = wv * 16 + t + 8 * half;
            float h = fmaxf(acc[t] * sc + sh, 0.f) * s_alpha[r];
            const int n = s_n[r];
            if (n != curn) {
                if (curn >= 0) atomicAdd(&fout[((long)curb * NN + curn) * OUT + o], run);
                curn = n; curb = s_b[r]; run = h;
            } else run += h;
        }
        atomicAdd(&fout[((long)curb * NN + curn) * OUT + o], run);
    }
}

// ---------------- head: per-point attention logits (wave per point) ----------------
__global__ __launch_bounds__(256) void k_head1(const float* __restrict__ f3, const float* __restrict__ Wa1,
                                               const float* __restrict__ ba1, const float* __restrict__ Wa2,
                                               const float* __restrict__ ba2, float* __restrict__ wlog) {
    __shared__ float s_row[8][256];
    const int tid = threadIdx.x, lane = tid & 31, wv = tid >> 5;
    const long P = (long)blockIdx.x * 8 + wv;
    const float* fr = f3 + P * 256;
    for (int i = lane; i < 256; i += 32) s_row[wv][i] = fr[i];
    __syncthreads();
    float part = 0.f;
#pragma unroll
    for (int rep = 0; rep < 2; ++rep) {
        const int o = lane + rep * 32;
        float acc = ba1[o];
        for (int i = 0; i < 256; ++i) acc = fmaf(s_row[wv][i], Wa1[i * 64 + o], acc);
        part += fmaxf(acc, 0.f) * Wa2[o];
    }
#pragma unroll
    for (int off = 16; off > 0; off >>= 1) part += __shfl_xor(part, off, 32);
    if (lane == 0) wlog[P] = part + ba2[0];
}

// ---------------- head: per-batch softmax over N, pool, FC ----------------
__global__ __launch_bounds__(256) void k_head2(const float* __restrict__ f3, const float* __restrict__ wlog,
                                               const float* __restrict__ Wfc, const float* __restrict__ bfc,
                                               float* __restrict__ out) {
    const int b = blockIdx.x, tid = threadIdx.x;
    __shared__ float red[256];
    __shared__ float s_aw[NN];
    __shared__ float s_pool[256];
    __shared__ float s_mx, s_z;
    const float* wl = wlog + (long)b * NN;
    float mx = -3.4e38f;
    for (int n = tid; n < NN; n += 256) mx = fmaxf(mx, wl[n]);
    red[tid] = mx; __syncthreads();
    for (int s = 128; s > 0; s >>= 1) { if (tid < s) red[tid] = fmaxf(red[tid], red[tid + s]); __syncthreads(); }
    if (tid == 0) s_mx = red[0];
    __syncthreads();
    float z = 0.f;
    for (int n = tid; n < NN; n += 256) { const float e = expf(wl[n] - s_mx); s_aw[n] = e; z += e; }
    red[tid] = z; __syncthreads();
    for (int s = 128; s > 0; s >>= 1) { if (tid < s) red[tid] += red[tid + s]; __syncthreads(); }
    if (tid == 0) s_z = red[0];
    __syncthreads();
    const float iz = 1.0f / s_z;
    for (int n = tid; n < NN; n += 256) s_aw[n] *= iz;
    __syncthreads();
    const int o = tid;
    float acc = 0.f;
    for (int n = 0; n < NN; ++n) acc = fmaf(f3[((long)b * NN + n) * 256 + o], s_aw[n], acc);
    s_pool[o] = acc;
    __syncthreads();
    float r = bfc[o];
    for (int i = 0; i < 256; ++i) r = fmaf(s_pool[i], Wfc[i * 256 + o], r);
    out[b * 256 + o] = fmaxf(r, 0.f);
}

// ---------------- host launcher ----------------
extern "C" void kernel_launch(void* const* d_in, const int* in_sizes, int n_in,
                              void* d_out, int out_size, void* d_ws, size_t ws_size,
                              hipStream_t stream) {
    (void)in_sizes; (void)n_in; (void)out_size; (void)ws_size;
    const float* x    = (const float*)d_in[0];
    const float* W1   = (const float*)d_in[1];
    const float* g1   = (const float*)d_in[2];
    const float* b1   = (const float*)d_in[3];
    const float* W2   = (const float*)d_in[4];
    const float* g2   = (const float*)d_in[5];
    const float* b2   = (const float*)d_in[6];
    const float* Ws2a = (const float*)d_in[7];
    const float* bs2a = (const float*)d_in[8];
    const float* Ws2b = (const float*)d_in[9];
    const float* bs2b = (const float*)d_in[10];
    const float* W3   = (const float*)d_in[11];
    const float* g3   = (const float*)d_in[12];
    const float* b3   = (const float*)d_in[13];
    const float* Ws3a = (const float*)d_in[14];
    const float* bs3a = (const float*)d_in[15];
    const float* Ws3b = (const float*)d_in[16];
    const float* bs3b = (const float*)d_in[17];
    const float* Wa1  = (const float*)d_in[18];
    const float* ba1  = (const float*)d_in[19];
    const float* Wa2  = (const float*)d_in[20];
    const float* ba2  = (const float*)d_in[21];
    const float* Wfc  = (const float*)d_in[22];
    const float* bfc  = (const float*)d_in[23];

    char* ws = (char*)d_ws;
    size_t off = 0;
    auto take = [&](size_t bytes) -> char* {
        char* p = ws + off;
        off += (bytes + 255) & ~(size_t)255;
        return p;
    };
    int*   idx11  = (int*)  take((size_t)BB * NN * 11 * 4);
    float* f1     = (float*)take((size_t)BB * NN * 64 * 4);
    float* f2     = (float*)take((size_t)BB * NN * 128 * 4);
    float* f3     = (float*)take((size_t)BB * NN * 256 * 4);
    float* logits = (float*)take((size_t)BB * NN * KNB * 4);
    float* alpha  = (float*)take((size_t)BB * NN * KNB * 4);
    float* stats1 = (float*)take(128 * 4);
    float* stats2 = (float*)take(256 * 4);
    float* stats3 = (float*)take(512 * 4);
    bf16*  W2b    = (bf16*) take(128 * 128 * 2);
    bf16*  Wa2b   = (bf16*) take(128 * 64 * 2);
    bf16*  W3b    = (bf16*) take(256 * 256 * 2);
    bf16*  Wa3b   = (bf16*) take(256 * 128 * 2);
    float* wlog   = (float*)take((size_t)BB * NN * 4);

    const long rowsE = (long)BB * NN * KNB;     // 327680 edge rows

    // zero accumulators
    k_zero<<<4096, 256, 0, stream>>>(f2, (long)BB * NN * 128);
    k_zero<<<8192, 256, 0, stream>>>(f3, (long)BB * NN * 256);
    k_zero<<<1, 128, 0, stream>>>(stats1, 128);
    k_zero<<<1, 256, 0, stream>>>(stats2, 256);
    k_zero<<<2, 256, 0, stream>>>(stats3, 512);

    // bf16 weight copies for WMMA
    k_cvt<<<(128 * 128 + 255) / 256, 256, 0, stream>>>(W2,   W2b,  128 * 128);
    k_cvt<<<(128 * 64  + 255) / 256, 256, 0, stream>>>(Ws2a, Wa2b, 128 * 64);
    k_cvt<<<(256 * 256 + 255) / 256, 256, 0, stream>>>(W3,   W3b,  256 * 256);
    k_cvt<<<(256 * 128 + 255) / 256, 256, 0, stream>>>(Ws3a, Wa3b, 256 * 128);

    // KNN
    k_knn<<<dim3(NN / 256, BB), 256, 0, stream>>>(x, idx11);

    // layer 1 (10 -> 64), two-pass BN + max over k
    k_l1_stats<<<(int)(rowsE / 256), 256, 0, stream>>>(x, idx11, W1, stats1);
    k_l1_apply<<<(BB * NN) / 256, 256, 0, stream>>>(x, idx11, W1, g1, b1, stats1, f1);

    // soft-edge layer 2 (e:128 -> 128, attn 128 -> 64 -> 1)
    k_soft_stats<128, 128, 64><<<(int)(rowsE / 64), 128, 0, stream>>>(
        f1, idx11, W2b, Wa2b, Ws2b, bs2a, bs2b, stats2, logits);
    k_softmax_k<<<(BB * NN) / 256, 256, 0, stream>>>(logits, alpha);
    k_soft_apply<128, 128><<<(int)(rowsE / 64), 128, 0, stream>>>(
        f1, idx11, W2b, stats2, g2, b2, alpha, f2);

    // soft-edge layer 3 (e:256 -> 256, attn 256 -> 128 -> 1)
    k_soft_stats<256, 256, 128><<<(int)(rowsE / 64), 128, 0, stream>>>(
        f2, idx11, W3b, Wa3b, Ws3b, bs3a, bs3b, stats3, logits);
    k_softmax_k<<<(BB * NN) / 256, 256, 0, stream>>>(logits, alpha);
    k_soft_apply<256, 256><<<(int)(rowsE / 64), 128, 0, stream>>>(
        f2, idx11, W3b, stats3, g3, b3, alpha, f3);

    // head: attention pooling + FC
    k_head1<<<(BB * NN) / 8, 256, 0, stream>>>(f3, Wa1, ba1, Wa2, ba2, wlog);
    k_head2<<<BB, 256, 0, stream>>>(f3, wlog, Wfc, bfc, (float*)d_out);
}
